// HomeostaticPredictiveMemory_71897752535458
// MI455X (gfx1250) — matrix operand order repaired and checked by reading, not compiled, not run on hardware
//
#include <hip/hip_runtime.h>
#include <hip/hip_bf16.h>
#include <math.h>

// ---------------- problem constants ----------------
constexpr int D  = 1024;   // d_model
constexpr int S  = 4;      // n_slots
constexpr int SD = 1024;   // slot_dim
constexpr int GH = 64;     // gate hidden
constexpr int SE = 8;      // state embed
constexpr int B  = 8;
constexpr int T  = 1024;
constexpr int W1ROWS = D + 1 + SE;   // 1033

// ---------------- WMMA fragment types ----------------
typedef __attribute__((ext_vector_type(16))) __bf16 v16bf;
typedef __attribute__((ext_vector_type(8)))  __bf16 v8bf;
typedef __attribute__((ext_vector_type(8)))  float  v8f;

union BF16x16 { v16bf v; v8bf h[2]; };

// A fragment (16x32 bf16, row-major source):
//   lane = half*16 + m  is WRONG; layout is: m = lane&15, half = lane>>4
//   elems 0..7  = A[m][kk + half*16 + 0..7]
//   elems 8..15 = A[m][kk + half*16 + 8..15]
// B fragment (32x16 bf16, from N-major (transposed) source BT[n][k]):
//   n = n0 + (lane&15); elems 0..15 = BT[n][kk + half*16 + 0..15]
// C/D fragment (16x16 f32): elem r = C[half*8 + r][lane&15]

__device__ __forceinline__ v8f wmma_bf16(const BF16x16& a, const BF16x16& b, v8f c) {
  return __builtin_amdgcn_wmma_f32_16x16x32_bf16(false, a.v, false, b.v,
                                                 (short)0, c, false, false);
}

// ---------------- prep kernels ----------------
__global__ void k_cvt(const float* __restrict__ src, __bf16* __restrict__ dst, int n) {
  for (int i = blockIdx.x * blockDim.x + threadIdx.x; i < n; i += gridDim.x * blockDim.x)
    dst[i] = (__bf16)src[i];
}

// predT[s][o][i] = pred_w[s][i][o]   (bf16)
__global__ void k_predT(const float* __restrict__ pred_w, __bf16* __restrict__ predT) {
  const int n = S * D * D;
  for (int idx = blockIdx.x * blockDim.x + threadIdx.x; idx < n; idx += gridDim.x * blockDim.x) {
    int s = idx / (D * D);
    int rem = idx - s * D * D;
    int o = rem / D;
    int i = rem - o * D;
    predT[idx] = (__bf16)pred_w[((size_t)s * D + i) * D + o];
  }
}

// writeT[s][o][i] = write_w[s][i][o]  (bf16), o < SD, i < D
__global__ void k_writeT(const float* __restrict__ write_w, __bf16* __restrict__ writeT) {
  const int n = S * SD * D;
  for (int idx = blockIdx.x * blockDim.x + threadIdx.x; idx < n; idx += gridDim.x * blockDim.x) {
    int s = idx / (SD * D);
    int rem = idx - s * SD * D;
    int o = rem / D;
    int i = rem - o * D;
    writeT[idx] = (__bf16)write_w[((size_t)s * D + i) * SD + o];
  }
}

// g1T[s*GH + j][k] = gate_w1[s][k][j], k < D   (bf16)
__global__ void k_g1T(const float* __restrict__ gate_w1, __bf16* __restrict__ g1T) {
  const int n = S * GH * D;
  for (int idx = blockIdx.x * blockDim.x + threadIdx.x; idx < n; idx += gridDim.x * blockDim.x) {
    int col = idx / D;          // 0..255
    int k   = idx - col * D;
    int s = col / GH, j = col - s * GH;
    g1T[idx] = (__bf16)gate_w1[((size_t)s * W1ROWS + k) * GH + j];
  }
}

__global__ void k_zinit(float* __restrict__ zbuf) {
  int idx = blockIdx.x * blockDim.x + threadIdx.x;   // b*S + s
  if (idx < B * S) {
    int b = idx / S, s = idx - b * S;
    zbuf[((size_t)b * T) * S + s] = 0.0f;            // z at t=0
  }
}

// ---------------- surprise / z kernel (WMMA + fused reduction) ----------------
// grid: (ttile=64, b=8, s=4), block 256 (8 waves). Rows t = t0..t0+15 predict t+1.
__global__ __launch_bounds__(256) void k_z(const __bf16* __restrict__ h_bf,
                                           const __bf16* __restrict__ predT,
                                           const float*  __restrict__ h_seq,
                                           const float*  __restrict__ pred_b,
                                           float* __restrict__ zbuf) {
  const int t0 = blockIdx.x * 16, b = blockIdx.y, s = blockIdx.z;
  __shared__ __bf16 atile[16 * D];
  __shared__ float err_lds[16];
  const int tid = threadIdx.x;
  {
    const v8bf* src = (const v8bf*)(h_bf + ((size_t)b * T + t0) * D);
    v8bf* dst = (v8bf*)atile;
    for (int i = tid; i < 16 * D / 8; i += 256) dst[i] = src[i];
  }
  if (tid < 16) err_lds[tid] = 0.0f;
  __syncthreads();

  const int lane = tid & 31, wave = tid >> 5;
  const int half = lane >> 4, l16 = lane & 15;

  float errAcc[8];
#pragma unroll
  for (int r = 0; r < 8; ++r) errAcc[r] = 0.0f;

  for (int i = 0; i < 8; ++i) {
    const int n = (wave + 8 * i) * 16 + l16;         // output column
    v8f acc = {};
    const __bf16* brow = predT + ((size_t)s * D + n) * D;
    for (int kk = 0; kk < D; kk += 32) {
      BF16x16 a, bm;
      const v8bf* ap = (const v8bf*)(atile + l16 * D + kk + half * 16);
      a.h[0] = ap[0]; a.h[1] = ap[1];
      const v8bf* bp = (const v8bf*)(brow + kk + half * 16);
      bm.h[0] = bp[0]; bm.h[1] = bp[1];
      acc = wmma_bf16(a, bm, acc);
    }
    const float pb = pred_b[s * D + n];
#pragma unroll
    for (int r = 0; r < 8; ++r) {
      const int t = t0 + half * 8 + r;
      if (t < T - 1) {
        const float target = h_seq[((size_t)b * T + t + 1) * D + n];
        const float diff = acc[r] + pb - target;
        errAcc[r] += diff * diff;
      }
    }
  }
#pragma unroll
  for (int r = 0; r < 8; ++r) atomicAdd(&err_lds[half * 8 + r], errAcc[r]);
  __syncthreads();
  if (tid < 16) {
    const int t = t0 + tid;
    if (t < T - 1)   // mu=0, sigma_safe=1 -> z = mean err
      zbuf[((size_t)b * T + t + 1) * S + s] = err_lds[tid] * (1.0f / D);
  }
}

// ---------------- gate hidden GEMM (8192 x 1024) x (1024 x 256) ----------------
// grid: (rowtiles=512, colgroup=4), block 128 (4 waves, one 16-col ntile each)
__global__ __launch_bounds__(128) void k_h1(const __bf16* __restrict__ h_bf,
                                            const __bf16* __restrict__ g1T,
                                            float* __restrict__ H1) {
  const int row0 = blockIdx.x * 16;
  const int tid = threadIdx.x, wave = tid >> 5, lane = tid & 31;
  const int half = lane >> 4, l16 = lane & 15;
  const int n = blockIdx.y * 64 + wave * 16 + l16;
  __shared__ __bf16 atile[16 * D];
  {
    const v8bf* src = (const v8bf*)(h_bf + (size_t)row0 * D);
    v8bf* dst = (v8bf*)atile;
    for (int i = tid; i < 16 * D / 8; i += 128) dst[i] = src[i];
  }
  __syncthreads();
  v8f acc = {};
  const __bf16* brow = g1T + (size_t)n * D;
  for (int kk = 0; kk < D; kk += 32) {
    BF16x16 a, bm;
    const v8bf* ap = (const v8bf*)(atile + l16 * D + kk + half * 16);
    a.h[0] = ap[0]; a.h[1] = ap[1];
    const v8bf* bp = (const v8bf*)(brow + kk + half * 16);
    bm.h[0] = bp[0]; bm.h[1] = bp[1];
    acc = wmma_bf16(a, bm, acc);
  }
#pragma unroll
  for (int r = 0; r < 8; ++r)
    H1[((size_t)row0 + half * 8 + r) * (S * GH) + n] = acc[r];
}

// ---------------- gate elementwise (state is provably always OPEN) ----------------
__global__ void k_gate(const float* __restrict__ H1,
                       const float* __restrict__ zbuf,
                       const float* __restrict__ gate_w1,
                       const float* __restrict__ gate_b1,
                       const float* __restrict__ gate_w2,
                       const float* __restrict__ gate_b2,
                       const float* __restrict__ state_embed,
                       float* __restrict__ gbuf) {
  const int idx = blockIdx.x * blockDim.x + threadIdx.x;   // (b*T+t)*S + s
  if (idx >= B * T * S) return;
  const int s = idx & (S - 1);
  const int bt = idx >> 2;
  const float z = zbuf[idx];
  float acc = 0.0f;
  for (int j = 0; j < GH; ++j) {
    float pre = H1[(size_t)bt * (S * GH) + s * GH + j];
    pre += z * gate_w1[((size_t)s * W1ROWS + D) * GH + j];     // z-row of W1
    float c = gate_b1[s * GH + j];
#pragma unroll
    for (int e = 0; e < SE; ++e)                               // emb = state_embed[0]
      c += state_embed[e] * gate_w1[((size_t)s * W1ROWS + D + 1 + e) * GH + j];
    pre += c;
    const float gh = 0.5f * pre * (1.0f + erff(pre * 0.70710678118654752f));
    acc += gh * gate_w2[s * GH + j];
  }
  const float logit = acc + gate_b2[s];
  float g = 1.0f / (1.0f + expf(-logit));                      // gain(OPEN)=1
  if (fabsf(z) > 3.0f) g = 1.0f;                               // min(g+1, 1)
  gbuf[idx] = g;
}

// ---------------- fused writes-GEMM + EMA scan ----------------
// grid: (chunk=SD/64=16, s=4, b=8), block 128 (4 waves).
// Dynamic LDS: write_wT slab 128KB + A-tile 32KB + x-tile 4KB + g 64B.
__global__ __launch_bounds__(128) void k_scan(const __bf16* __restrict__ h_bf,
                                              const __bf16* __restrict__ writeT,
                                              const float*  __restrict__ write_b,
                                              const float*  __restrict__ gbuf,
                                              const float*  __restrict__ w0,
                                              float* __restrict__ out) {
  extern __shared__ char smem[];
  __bf16* wlds  = (__bf16*)smem;                               // 64*D
  __bf16* atile = (__bf16*)(smem + 64 * D * 2);                // 16*D
  float*  xtile = (float*)(smem + 64 * D * 2 + 16 * D * 2);    // 16*64
  float*  glds  = xtile + 16 * 64;                             // 16

  const int n0 = blockIdx.x * 64, s = blockIdx.y, b = blockIdx.z;
  const int tid = threadIdx.x, wave = tid >> 5, lane = tid & 31;
  const int half = lane >> 4, l16 = lane & 15;

  { // stage the 64-column weight slab once (contiguous in writeT)
    const v8bf* src = (const v8bf*)(writeT + ((size_t)s * SD + n0) * D);
    v8bf* dst = (v8bf*)wlds;
    for (int i = tid; i < 64 * D / 8; i += 128) dst[i] = src[i];
  }
  float wcar = 0.0f, wb = 0.0f;
  if (tid < 64) {
    wcar = w0[s * SD + n0 + tid];
    wb   = write_b[s * SD + n0 + tid];
  }
  __syncthreads();

  for (int tt = 0; tt < T / 16; ++tt) {
    const int t0 = tt * 16;
    { // stage 16 rows of h (contiguous)
      const v8bf* src = (const v8bf*)(h_bf + ((size_t)b * T + t0) * D);
      v8bf* dst = (v8bf*)atile;
      for (int i = tid; i < 16 * D / 8; i += 128) dst[i] = src[i];
    }
    if (tid < 16) glds[tid] = gbuf[((size_t)b * T + t0 + tid) * S + s];
    __syncthreads();

    v8f acc = {};
    const __bf16* brow = wlds + (size_t)(wave * 16 + l16) * D;
    for (int kk = 0; kk < D; kk += 32) {
      BF16x16 a, bm;
      const v8bf* ap = (const v8bf*)(atile + l16 * D + kk + half * 16);
      a.h[0] = ap[0]; a.h[1] = ap[1];
      const v8bf* bp = (const v8bf*)(brow + kk + half * 16);
      bm.h[0] = bp[0]; bm.h[1] = bp[1];
      acc = wmma_bf16(a, bm, acc);
    }
#pragma unroll
    for (int r = 0; r < 8; ++r)
      xtile[(half * 8 + r) * 64 + wave * 16 + l16] = acc[r];
    __syncthreads();

    if (tid < 64) {   // 16-step EMA down the tile, stream out
      float w = wcar;
#pragma unroll 1
      for (int q = 0; q < 16; ++q) {
        const float g = glds[q];
        const float x = xtile[q * 64 + tid] + wb;
        w = (1.0f - g) * w + g * x;
        out[((size_t)b * T + t0 + q) * (S * SD) + (size_t)s * SD + n0 + tid] = w;
      }
      wcar = w;
    }
    __syncthreads();
  }
}

// ---------------- launcher ----------------
extern "C" void kernel_launch(void* const* d_in, const int* in_sizes, int n_in,
                              void* d_out, int out_size, void* d_ws, size_t ws_size,
                              hipStream_t stream) {
  const float* h_seq       = (const float*)d_in[0];
  const float* pred_w      = (const float*)d_in[1];
  const float* pred_b      = (const float*)d_in[2];
  const float* write_w     = (const float*)d_in[3];
  const float* write_b     = (const float*)d_in[4];
  const float* gate_w1     = (const float*)d_in[5];
  const float* gate_b1     = (const float*)d_in[6];
  const float* gate_w2     = (const float*)d_in[7];
  const float* gate_b2     = (const float*)d_in[8];
  const float* state_embed = (const float*)d_in[9];
  const float* w0          = (const float*)d_in[10];
  float* out = (float*)d_out;

  // workspace layout (bytes)
  char* ws = (char*)d_ws;
  __bf16* h_bf   = (__bf16*)(ws);                         // 16 MB
  __bf16* predT  = (__bf16*)(ws + (16u << 20));           //  8 MB
  __bf16* writeT = (__bf16*)(ws + (24u << 20));           //  8 MB
  __bf16* g1T    = (__bf16*)(ws + (32u << 20));           // 512 KB
  float*  zbuf   = (float*)(ws + (33u << 20));            // 128 KB
  float*  gbuf   = (float*)(ws + (34u << 20));            // 128 KB
  float*  H1     = (float*)(ws + (35u << 20));            //  8 MB

  // ---- prep ----
  k_cvt   <<<4096, 256, 0, stream>>>(h_seq, h_bf, B * T * D);
  k_predT <<<4096, 256, 0, stream>>>(pred_w, predT);
  k_writeT<<<4096, 256, 0, stream>>>(write_w, writeT);
  k_g1T   <<<1024, 256, 0, stream>>>(gate_w1, g1T);
  k_zinit <<<1, 64, 0, stream>>>(zbuf);

  // ---- surprise z ----
  k_z<<<dim3(T / 16, B, S), 256, 0, stream>>>(h_bf, predT, h_seq, pred_b, zbuf);

  // ---- gate hidden + gate ----
  k_h1<<<dim3(B * T / 16, (S * GH) / 64), 128, 0, stream>>>(h_bf, g1T, H1);
  k_gate<<<(B * T * S + 255) / 256, 256, 0, stream>>>(H1, zbuf, gate_w1, gate_b1,
                                                      gate_w2, gate_b2, state_embed, gbuf);

  // ---- fused writes GEMM + scan ----
  const size_t smem = (size_t)(64 * D * 2 + 16 * D * 2 + 16 * 64 * 4 + 16 * 4);
  k_scan<<<dim3(SD / 64, S, B), 128, smem, stream>>>(h_bf, writeT, write_b,
                                                     gbuf, w0, out);
}
